// ViT_Reconstruction_Model_88708254531963
// MI455X (gfx1250) — matrix operand order repaired and checked by reading, not compiled, not run on hardware
//
#include <hip/hip_runtime.h>
#include <math.h>

typedef _Float16 v16h __attribute__((ext_vector_type(16)));
typedef _Float16 v8h  __attribute__((ext_vector_type(8)));
typedef float    v8f  __attribute__((ext_vector_type(8)));
typedef int      v4i  __attribute__((ext_vector_type(4)));

#define TOKENS 1568
#define NTOK   196
#define BATCH  8

#if __has_builtin(__builtin_amdgcn_global_load_async_to_lds_b128)
#define HAVE_ASYNC_LDS 1
typedef __attribute__((address_space(1))) v4i* gv4i_p;   // global int4*
typedef __attribute__((address_space(3))) v4i* lv4i_p;   // LDS int4*
#else
#define HAVE_ASYNC_LDS 0
#endif

__device__ __forceinline__ float gelu_t(float x) {
  float x3 = x * x * x;
  return 0.5f * x * (1.0f + tanhf(0.7978845608028654f * (x + 0.044715f * x3)));
}

// =======================================================================
// Block-tiled WMMA GEMM.  C[M,N] = act(A[M,K] @ B + bias) (+ resid)
//   block tile: 32 (M) x 128 (N), BK = 64, 256 threads = 8 waves (2x4)
//   each wave: 16x32 sub-tile -> 2 accumulators, 4 WMMA per K-iteration
//   A staged to LDS as f32 via async global->LDS copies (ASYNCcnt path)
//   B staged to LDS as f16, transposed to [col][k] for contiguous frags
// transB==0: B(k,n) = W[k*N+n] ; transB==1: B(k,n) = W[n*K+k]
// Requires: M%32==0, N%128==0, K%64==0 (true for every GEMM in this model)
// =======================================================================
__global__ __launch_bounds__(256)
void gemm_wmma_kernel(const float* __restrict__ A, const float* __restrict__ W,
                      const float* __restrict__ bias, const float* __restrict__ resid,
                      float* __restrict__ C, int M, int K, int N, int act, int transB)
{
  __shared__ float    sAf[32][64];     // A tile, f32 (async-copied raw)
  __shared__ _Float16 sBt[128][64];    // B tile, f16, [col][k]

  const int tid  = threadIdx.x;
  const int lane = tid & 31;
  const int w    = tid >> 5;
  const int wm   = w & 1;              // 0..1 : 16-row slice
  const int wn   = w >> 1;             // 0..3 : 32-col slice
  const int hi   = lane >> 4;          // lane half
  const int l15  = lane & 15;
  const int kgA  = hi * 16;            // A fragment K-group (contiguous layout)
  const int kgB  = hi * 16;            // B fragment K-group
  const int bm0  = blockIdx.y * 32;
  const int bn0  = blockIdx.x * 128;

  // staging maps
  const int arow = tid >> 3;           // 0..31
  const int acg  = (tid & 7) * 8;      // 0..56, 8 floats per thread
  const int bkN  = tid >> 2;           // non-trans: k row 0..63
  const int bngN = (tid & 3) * 32;     // non-trans: 32 cols per thread
  const int bcT  = tid >> 1;           // trans: col 0..127
  const int bkgT = (tid & 1) * 32;     // trans: 32 k per thread

  v8f acc0 = {}, acc1 = {};
  const int rowL = wm * 16 + l15;      // A row within tile held by this lane
  const int c0   = wn * 32 + l15;      // B col for sub-tile 0
  const int c1   = wn * 32 + 16 + l15; // B col for sub-tile 1

  for (int k0 = 0; k0 < K; k0 += 64) {
    // ---- stage A (32x64 f32) : one 16B async copy x2 per thread ----
    {
      const float* src = A + (size_t)(bm0 + arow) * K + k0 + acg;
#if HAVE_ASYNC_LDS
      __builtin_amdgcn_global_load_async_to_lds_b128(
          (gv4i_p)src, (lv4i_p)&sAf[arow][acg], 0, 0);
      __builtin_amdgcn_global_load_async_to_lds_b128(
          (gv4i_p)src, (lv4i_p)&sAf[arow][acg], 16, 0);
#else
      float4 v0 = *(const float4*)src;
      float4 v1 = *(const float4*)(src + 4);
      *(float4*)&sAf[arow][acg]     = v0;
      *(float4*)&sAf[arow][acg + 4] = v1;
#endif
    }
    // ---- stage B (64x128 -> f16 [col][k]) ----
    if (transB) {
      const float* src = W + (size_t)(bn0 + bcT) * K + k0 + bkgT;
      __builtin_prefetch(src + 64, 0, 1);
#pragma unroll
      for (int j = 0; j < 32; ++j) sBt[bcT][bkgT + j] = (_Float16)src[j];
    } else {
      const float* src = W + (size_t)(k0 + bkN) * N + bn0 + bngN;
      __builtin_prefetch(src + (size_t)64 * N, 0, 1);
#pragma unroll
      for (int j = 0; j < 32; ++j) sBt[bngN + j][bkN] = (_Float16)src[j];
    }
#if HAVE_ASYNC_LDS
    __builtin_amdgcn_s_wait_asynccnt(0);
#endif
    __syncthreads();

    // ---- 2 K-steps of 32, 2 N-tiles -> 4 WMMA ----
#pragma unroll
    for (int s = 0; s < 2; ++s) {
      const int ks = s * 32;
      v16h a;
#pragma unroll
      for (int j = 0; j < 8; ++j) a[j]     = (_Float16)sAf[rowL][ks + kgA + j];
#pragma unroll
      for (int j = 0; j < 8; ++j) a[8 + j] = (_Float16)sAf[rowL][ks + 8 + kgA + j];

      v8h b0lo = *(const v8h*)&sBt[c0][ks + kgB];
      v8h b0hi = *(const v8h*)&sBt[c0][ks + kgB + 8];
      v16h b0 = __builtin_shufflevector(b0lo, b0hi, 0,1,2,3,4,5,6,7,8,9,10,11,12,13,14,15);
      v8h b1lo = *(const v8h*)&sBt[c1][ks + kgB];
      v8h b1hi = *(const v8h*)&sBt[c1][ks + kgB + 8];
      v16h b1 = __builtin_shufflevector(b1lo, b1hi, 0,1,2,3,4,5,6,7,8,9,10,11,12,13,14,15);

      acc0 = __builtin_amdgcn_wmma_f32_16x16x32_f16(false, a, false, b0, (short)0, acc0, false, false);
      acc1 = __builtin_amdgcn_wmma_f32_16x16x32_f16(false, a, false, b1, (short)0, acc1, false, false);
    }
    __syncthreads();
  }

  // ---- epilogue: bias / gelu / residual, direct f32 stores ----
#pragma unroll
  for (int t = 0; t < 2; ++t) {
    v8f acc = t ? acc1 : acc0;
    int col = bn0 + wn * 32 + t * 16 + l15;
    float bv = bias ? bias[col] : 0.0f;
#pragma unroll
    for (int r = 0; r < 8; ++r) {
      int row = bm0 + wm * 16 + r + hi * 8;
      float v = acc[r] + bv;
      if (act == 1) v = gelu_t(v);
      size_t o = (size_t)row * N + col;
      if (resid) v += resid[o];
      C[o] = v;
    }
  }
}

// ---------------- LayerNorm ----------------
__global__ __launch_bounds__(256)
void layernorm_kernel(const float* __restrict__ x, float* __restrict__ y,
                      const float* __restrict__ g, const float* __restrict__ bb, int dim)
{
  __shared__ float red[256];
  int row = blockIdx.x;
  const float* xr = x + (size_t)row * dim;
  float lv[4];
  int cnt = 0;
  float s = 0.f;
  for (int i = threadIdx.x; i < dim; i += 256) { float v = xr[i]; lv[cnt++] = v; s += v; }
  red[threadIdx.x] = s; __syncthreads();
  for (int o = 128; o > 0; o >>= 1) { if (threadIdx.x < o) red[threadIdx.x] += red[threadIdx.x + o]; __syncthreads(); }
  float mean = red[0] / dim; __syncthreads();
  s = 0.f;
  { int c = 0; for (int i = threadIdx.x; i < dim; i += 256) { float d = lv[c++] - mean; s += d * d; } }
  red[threadIdx.x] = s; __syncthreads();
  for (int o = 128; o > 0; o >>= 1) { if (threadIdx.x < o) red[threadIdx.x] += red[threadIdx.x + o]; __syncthreads(); }
  float inv = rsqrtf(red[0] / dim + 1e-6f);
  float* yr = y + (size_t)row * dim;
  int c = 0;
  for (int i = threadIdx.x; i < dim; i += 256) yr[i] = (lv[c++] - mean) * inv * g[i] + bb[i];
}

// ---------------- sincos positional add ----------------
__global__ void addpos_kernel(float* x, int dim) {
  int i = blockIdx.x * blockDim.x + threadIdx.x;
  if (i >= TOKENS * dim) return;
  int t = i / dim, f = i % dim;
  int pos = t % NTOK;
  float ang = pos * __expf(-logf(10000.f) * (2.0f * (f / 2)) / dim);
  x[i] += (f & 1) ? cosf(ang) : sinf(ang);
}

// ---------------- fused softmax attention (per b,h,query) ----------------
__global__ __launch_bounds__(256)
void attention_kernel(const float* __restrict__ qkv, float* __restrict__ out, int H, int dh)
{
  int n = blockIdx.x, h = blockIdx.y, b = blockIdx.z;
  int D = H * dh;
  __shared__ float sc[NTOK];
  __shared__ float qv[64];
  __shared__ float red[256];
  const float* qp = qkv + (size_t)(b * NTOK + n) * 3 * D + h * dh;
  for (int d = threadIdx.x; d < dh; d += 256) qv[d] = qp[d];
  __syncthreads();
  float scale = rsqrtf((float)dh);
  for (int m = threadIdx.x; m < NTOK; m += 256) {
    const float* kp = qkv + (size_t)(b * NTOK + m) * 3 * D + D + h * dh;
    float s = 0.f;
    for (int d = 0; d < dh; ++d) s += qv[d] * kp[d];
    sc[m] = s * scale;
  }
  __syncthreads();
  float mx = -1e30f;
  for (int m = threadIdx.x; m < NTOK; m += 256) mx = fmaxf(mx, sc[m]);
  red[threadIdx.x] = mx; __syncthreads();
  for (int o = 128; o > 0; o >>= 1) { if (threadIdx.x < o) red[threadIdx.x] = fmaxf(red[threadIdx.x], red[threadIdx.x + o]); __syncthreads(); }
  mx = red[0]; __syncthreads();
  float s = 0.f;
  for (int m = threadIdx.x; m < NTOK; m += 256) { float e = __expf(sc[m] - mx); sc[m] = e; s += e; }
  red[threadIdx.x] = s; __syncthreads();
  for (int o = 128; o > 0; o >>= 1) { if (threadIdx.x < o) red[threadIdx.x] += red[threadIdx.x + o]; __syncthreads(); }
  float inv = 1.0f / red[0]; __syncthreads();
  for (int d = threadIdx.x; d < dh; d += 256) {
    float acc = 0.f;
    for (int m = 0; m < NTOK; ++m)
      acc += sc[m] * qkv[(size_t)(b * NTOK + m) * 3 * D + 2 * D + h * dh + d];
    out[(size_t)(b * NTOK + n) * D + h * dh + d] = acc * inv;
  }
}

// ---------------- patchify / unpatchify ----------------
__global__ void patchify_kernel(const float* __restrict__ img, float* __restrict__ xp) {
  int i = blockIdx.x * blockDim.x + threadIdx.x;
  if (i >= TOKENS * 768) return;
  int t = i / 768, f = i % 768;
  int b = t / NTOK, tt = t % NTOK;
  int gh = tt / 14, gw = tt % 14;
  int c = f / 256, r = f % 256, ph = r / 16, pw = r % 16;
  xp[i] = img[((size_t)(b * 3 + c) * 224 + gh * 16 + ph) * 224 + gw * 16 + pw];
}

__global__ void unpatch_kernel(const float* __restrict__ px, float* __restrict__ recon) {
  int i = blockIdx.x * blockDim.x + threadIdx.x;
  if (i >= BATCH * 3 * 224 * 224) return;
  int b = i / (3 * 224 * 224); int rr = i % (3 * 224 * 224);
  int c = rr / (224 * 224); int r2 = rr % (224 * 224);
  int y = r2 / 224, x = r2 % 224;
  int gh = y / 16, ph = y % 16, gw = x / 16, pw = x % 16;
  recon[i] = px[(size_t)(b * NTOK + gh * 14 + gw) * 768 + c * 256 + ph * 16 + pw];
}

// ---------------- FIM head (128 -> 1) ----------------
__global__ void fimhead_kernel(const float* __restrict__ f, const float* __restrict__ w,
                               const float* __restrict__ b, float* __restrict__ out) {
  int t = blockIdx.x * blockDim.x + threadIdx.x;
  if (t >= TOKENS) return;
  float s = b[0];
  const float* fr = f + (size_t)t * 128;
  for (int k = 0; k < 128; ++k) s += fr[k] * w[k];
  out[t] = s;
}

// ---------------- VQ helpers ----------------
__global__ void cbnorm_kernel(const float* __restrict__ cb, float* __restrict__ out, int Kcb) {
  int k = blockIdx.x * blockDim.x + threadIdx.x;
  if (k >= Kcb) return;
  float s = 0.f; const float* r = cb + (size_t)k * 512;
  for (int d = 0; d < 512; ++d) s += r[d] * r[d];
  out[k] = s;
}

__global__ __launch_bounds__(256)
void argmin_kernel(const float* __restrict__ S, const float* __restrict__ cbn,
                   int* __restrict__ idx, int Kcb) {
  __shared__ float rv[256]; __shared__ int ri[256];
  int t = blockIdx.x;
  float best = 1e30f; int bi = 0;
  for (int k = threadIdx.x; k < Kcb; k += 256) {
    float v = cbn[k] - 2.0f * S[(size_t)t * Kcb + k];
    if (v < best) { best = v; bi = k; }
  }
  rv[threadIdx.x] = best; ri[threadIdx.x] = bi; __syncthreads();
  for (int o = 128; o > 0; o >>= 1) {
    if (threadIdx.x < o) {
      if (rv[threadIdx.x + o] < rv[threadIdx.x] ||
          (rv[threadIdx.x + o] == rv[threadIdx.x] && ri[threadIdx.x + o] < ri[threadIdx.x])) {
        rv[threadIdx.x] = rv[threadIdx.x + o]; ri[threadIdx.x] = ri[threadIdx.x + o];
      }
    }
    __syncthreads();
  }
  if (threadIdx.x == 0) idx[t] = ri[0];
}

__global__ __launch_bounds__(256)
void vqfinal_kernel(const float* __restrict__ z, const float* __restrict__ cbH,
                    const float* __restrict__ cbL, const int* __restrict__ idxH,
                    const int* __restrict__ idxL, const float* __restrict__ logits,
                    float* __restrict__ finalq, float* __restrict__ lsum) {
  __shared__ float red[256];
  int t = blockIdx.x;
  float score = 1.0f / (1.0f + __expf(-logits[t]));
  bool hi = score > 0.6f;
  const float* cb = hi ? (cbH + (size_t)idxH[t] * 512) : (cbL + (size_t)idxL[t] * 512);
  const float* zr = z + (size_t)t * 512;
  float* fr = finalq + (size_t)t * 512;
  float s = 0.f;
  for (int d = threadIdx.x; d < 512; d += 256) {
    float q = cb[d]; float df = q - zr[d]; s += df * df; fr[d] = q;
  }
  red[threadIdx.x] = s; __syncthreads();
  for (int o = 128; o > 0; o >>= 1) { if (threadIdx.x < o) red[threadIdx.x] += red[threadIdx.x + o]; __syncthreads(); }
  if (threadIdx.x == 0) lsum[t] = 1.25f * (red[0] / 512.0f) * score;   // (1+COMMIT)*msd*score
}

__global__ __launch_bounds__(256)
void lossreduce_kernel(const float* __restrict__ lsum, float* __restrict__ out) {
  __shared__ float red[256];
  float s = 0.f;
  for (int i = threadIdx.x; i < TOKENS; i += 256) s += lsum[i];
  red[threadIdx.x] = s; __syncthreads();
  for (int o = 128; o > 0; o >>= 1) { if (threadIdx.x < o) red[threadIdx.x] += red[threadIdx.x + o]; __syncthreads(); }
  if (threadIdx.x == 0) out[0] = red[0] / TOKENS;
}

// ---------------- Rayleigh channel: y = x + n * H^-1 ----------------
__device__ __forceinline__ unsigned hashu(unsigned x) {
  x ^= x >> 16; x *= 0x7feb352du; x ^= x >> 15; x *= 0x846ca68bu; x ^= x >> 16; return x;
}
__device__ __forceinline__ float u01(unsigned h) { return (h + 1u) * 2.3283064e-10f; }

__global__ void rayleigh_kernel(const float* __restrict__ x, float* __restrict__ y, int npairs) {
  int i = blockIdx.x * blockDim.x + threadIdx.x;
  if (i >= npairs) return;
  float hu1 = u01(hashu(0x1234567u));
  float hu2 = u01(hashu(0x89abcdeu));
  float rr = sqrtf(-2.0f * logf(hu1)) * 0.70710678f;
  float hr = rr * cosf(6.2831853f * hu2);
  float hi_ = rr * sinf(6.2831853f * hu2);
  float det = hr * hr + hi_ * hi_;
  float u1 = u01(hashu(2u * (unsigned)i + 101u));
  float u2 = u01(hashu(2u * (unsigned)i + 202u));
  float r = sqrtf(-2.0f * logf(u1)) * 0.031622776f;   // sqrt(n_var = 1e-3)
  float nr = r * cosf(6.2831853f * u2), ni = r * sinf(6.2831853f * u2);
  y[2 * i]     = x[2 * i]     + (nr * hr - ni * hi_) / det;
  y[2 * i + 1] = x[2 * i + 1] + (nr * hi_ + ni * hr) / det;
}

// =========================================================================
extern "C" void kernel_launch(void* const* d_in, const int* in_sizes, int n_in,
                              void* d_out, int out_size, void* d_ws, size_t ws_size,
                              hipStream_t stream)
{
  (void)in_sizes; (void)n_in; (void)out_size; (void)ws_size;
  auto in = [&](int i) { return (const float*)d_in[i]; };
  float* ws = (float*)d_ws;

  // workspace layout (floats)
  size_t oXP  = 0;
  size_t oX   = oXP  + (size_t)TOKENS * 768;
  size_t oLN  = oX   + (size_t)TOKENS * 768;
  size_t oQKV = oLN  + (size_t)TOKENS * 768;
  size_t oAO  = oQKV + (size_t)TOKENS * 2304;
  size_t oHID = oAO  + (size_t)TOKENS * 768;
  size_t oF   = oHID + (size_t)TOKENS * 3072;
  size_t oZ   = oF   + (size_t)TOKENS * 128;
  size_t oFIN = oZ   + (size_t)TOKENS * 512;
  size_t oY   = oFIN + (size_t)TOKENS * 512;
  size_t oD   = oY   + (size_t)TOKENS * 512;
  size_t oLS  = oD   + (size_t)TOKENS * 256;
  size_t oCBN = oLS  + TOKENS;
  size_t oIDX = oCBN + 4096 + 256;
  float* XP = ws + oXP;  float* X  = ws + oX;   float* LN  = ws + oLN;
  float* QKV = ws + oQKV; float* AO = ws + oAO; float* HID = ws + oHID;
  float* F  = ws + oF;   float* Z  = ws + oZ;   float* FIN = ws + oFIN;
  float* Y  = ws + oY;   float* Dd = ws + oD;   float* LS  = ws + oLS;
  float* CBNH = ws + oCBN; float* CBNL = CBNH + 4096;
  int* IDXH = (int*)(ws + oIDX); int* IDXL = IDXH + TOKENS;
  float* S = ws + oQKV;  // VQ scores alias QKV..HID region (idle during VQ)

  float* recon  = (float*)d_out;
  float* lossp  = recon + (size_t)BATCH * 3 * 224 * 224;
  float* fimlog = lossp + 1;

  auto gemm = [&](const float* A, const float* W, const float* bias, const float* resid,
                  float* C, int M, int K, int N, int act, int transB) {
    dim3 g(N / 128, M / 32);
    gemm_wmma_kernel<<<g, 256, 0, stream>>>(A, W, bias, resid, C, M, K, N, act, transB);
  };
  auto lnorm = [&](const float* x, float* y, int pg, int dim) {
    layernorm_kernel<<<TOKENS, 256, 0, stream>>>(x, y, in(pg), in(pg + 1), dim);
  };
  auto runblock = [&](float* x, int base, int dim, int H) {
    lnorm(x, LN, base + 0, dim);
    gemm(LN, in(base + 2), in(base + 3), nullptr, QKV, TOKENS, dim, 3 * dim, 0, 0);
    attention_kernel<<<dim3(NTOK, H, BATCH), 256, 0, stream>>>(QKV, AO, H, dim / H);
    gemm(AO, in(base + 4), in(base + 5), x, x, TOKENS, dim, dim, 0, 0);
    lnorm(x, LN, base + 6, dim);
    gemm(LN, in(base + 8), in(base + 9), nullptr, HID, TOKENS, dim, 4 * dim, 1, 0);
    gemm(HID, in(base + 10), in(base + 11), x, x, TOKENS, 4 * dim, dim, 0, 0);
  };

  // 1) patchify + patch embed + pos
  patchify_kernel<<<(TOKENS * 768 + 255) / 256, 256, 0, stream>>>(in(0), XP);
  gemm(XP, in(1), in(2), nullptr, X, TOKENS, 768, 768, 0, 0);
  addpos_kernel<<<(TOKENS * 768 + 255) / 256, 256, 0, stream>>>(X, 768);
  // 2) encoder (12 blocks, d=768, 12 heads) + final LN (in place)
  for (int i = 0; i < 12; ++i) runblock(X, 3 + i * 12, 768, 12);
  lnorm(X, X, 147, 768);
  // 3) FIM branch (d=128, 2 blocks, 4 heads) -> per-token logit
  gemm(X, in(149), in(150), nullptr, F, TOKENS, 768, 128, 0, 0);
  addpos_kernel<<<(TOKENS * 128 + 255) / 256, 256, 0, stream>>>(F, 128);
  for (int j = 0; j < 2; ++j) runblock(F, 151 + j * 12, 128, 4);
  lnorm(F, LN, 175, 128);
  fimhead_kernel<<<(TOKENS + 255) / 256, 256, 0, stream>>>(LN, in(177), in(178), fimlog);
  // 4) to_ch projection + q_norm (in place)
  gemm(X, in(179), in(180), nullptr, Z, TOKENS, 768, 512, 0, 0);
  lnorm(Z, Z, 181, 512);
  // 5) VQ: scores = z @ cb^T (WMMA), argmin with codebook norms, route by FIM score
  cbnorm_kernel<<<(4096 + 255) / 256, 256, 0, stream>>>(in(183), CBNH, 4096);
  cbnorm_kernel<<<1, 256, 0, stream>>>(in(184), CBNL, 256);
  gemm(Z, in(183), nullptr, nullptr, S, TOKENS, 512, 4096, 0, 1);
  argmin_kernel<<<TOKENS, 256, 0, stream>>>(S, CBNH, IDXH, 4096);
  gemm(Z, in(184), nullptr, nullptr, S, TOKENS, 512, 256, 0, 1);
  argmin_kernel<<<TOKENS, 256, 0, stream>>>(S, CBNL, IDXL, 256);
  vqfinal_kernel<<<TOKENS, 256, 0, stream>>>(Z, in(183), in(184), IDXH, IDXL, fimlog, FIN, LS);
  lossreduce_kernel<<<1, 256, 0, stream>>>(LS, lossp);
  // 6) Rayleigh channel (equalized: x + n*H^-1)
  rayleigh_kernel<<<(TOKENS * 256 + 255) / 256, 256, 0, stream>>>(FIN, Y, TOKENS * 256);
  // 7) decoder (d=256, 4 blocks, 4 heads) + head + unpatchify
  gemm(Y, in(185), in(186), nullptr, Dd, TOKENS, 512, 256, 0, 0);
  addpos_kernel<<<(TOKENS * 256 + 255) / 256, 256, 0, stream>>>(Dd, 256);
  for (int m = 0; m < 4; ++m) runblock(Dd, 187 + m * 12, 256, 4);
  lnorm(Dd, Dd, 235, 256);
  gemm(Dd, in(237), in(238), nullptr, XP, TOKENS, 256, 768, 0, 0);
  unpatch_kernel<<<(BATCH * 3 * 224 * 224 + 255) / 256, 256, 0, stream>>>(XP, recon);
}
